// PCILTConv2d_70892730187913
// MI455X (gfx1250) — 8-bit matrix operand order repaired and checked by reading, compiled, not run on hardware
//
#include <hip/hip_runtime.h>
#include <stdint.h>

typedef __attribute__((ext_vector_type(8))) int v8i;

constexpr int kB    = 8;
constexpr int kCin  = 128;
constexpr int kCout = 128;
constexpr int kH    = 56;
constexpr int kW    = 56;
constexpr int kHW   = kH * kW;
constexpr int kNpix = kB * kHW;
constexpr int kK    = kCin * 9;
constexpr int kSteps = kK / 64;
constexpr int kMTiles = kCout / 16;

__global__ void quant_x_kernel(const float* __restrict__ x,
                               uint32_t* __restrict__ xq_packed, int n4) {
    int i = blockIdx.x * blockDim.x + threadIdx.x;
    if (i >= n4) return;
    float4 v = reinterpret_cast<const float4*>(x)[i];
    uint32_t r = 0;
    float a;
    a = fminf(fmaxf(rintf(v.x * 255.0f), 0.0f), 255.0f); r |= (uint32_t)a;
    a = fminf(fmaxf(rintf(v.y * 255.0f), 0.0f), 255.0f); r |= (uint32_t)a << 8;
    a = fminf(fmaxf(rintf(v.z * 255.0f), 0.0f), 255.0f); r |= (uint32_t)a << 16;
    a = fminf(fmaxf(rintf(v.w * 255.0f), 0.0f), 255.0f); r |= (uint32_t)a << 24;
    xq_packed[i] = r;
}

__global__ void pack_w_kernel(const float* __restrict__ pcilt,
                              uint32_t* __restrict__ apack) {
    int d = blockIdx.x * blockDim.x + threadIdx.x;
    if (d >= kMTiles * kSteps * 32 * 8) return;
    int v     = d & 7;
    int lane  = (d >> 3) & 31;
    int g     = d >> 8;
    int kstep = g % kSteps;
    int mtile = g / kSteps;
    int m     = mtile * 16 + (lane & 15);
    int kbase = kstep * 64 + ((v >> 1) * 16) + ((v & 1) * 4) + ((lane & 16) ? 8 : 0);
    uint32_t r = 0;
#pragma unroll
    for (int b = 0; b < 4; ++b) {
        int kg  = kbase + b;
        int cin = kg / 9;
        int rr  = kg - cin * 9;
        int kh  = rr / 3;
        int kw  = rr - kh * 3;
        size_t idx = ((((size_t)m * kCin + cin) * 3 + kh) * 3 + kw) * 256 + 1;
        uint32_t q = (uint32_t)(pcilt[idx] + 0.5f);
        r |= (q & 0xFFu) << (8 * b);
    }
    apack[d] = r;
}

__global__ __launch_bounds__(256)
void conv_wmma_kernel(const uint8_t* __restrict__ xq,
                      const uint32_t* __restrict__ apack,
                      const float* __restrict__ bias,
                      float* __restrict__ out) {
    __shared__ __attribute__((aligned(32))) uint32_t ldsB[kSteps * 256];

    const int tid   = threadIdx.x;
    const int ntile = blockIdx.x;

    {
        int lane = tid >> 3;
        int v    = tid & 7;
        int n    = lane & 15;
        int p    = ntile * 16 + n;
        int bb   = p / kHW;
        int rem  = p - bb * kHW;
        int h    = rem / kW;
        int w    = rem - h * kW;
        int kb0  = ((v >> 1) * 16) + ((v & 1) * 4) + ((lane & 16) ? 8 : 0);
        for (int ks = 0; ks < kSteps; ++ks) {
            int kbase = ks * 64 + kb0;
            uint32_t r = 0;
#pragma unroll
            for (int b = 0; b < 4; ++b) {
                int kg  = kbase + b;
                int cin = kg / 9;
                int rr  = kg - cin * 9;
                int kh  = rr / 3;
                int kw2 = rr - kh * 3;
                int ih  = h + kh - 1;
                int iw  = w + kw2 - 1;
                uint32_t q = 0;
                if ((unsigned)ih < (unsigned)kH && (unsigned)iw < (unsigned)kW)
                    q = xq[(((size_t)bb * kCin + cin) * kH + ih) * kW + iw];
                r |= q << (8 * b);
            }
            ldsB[ks * 256 + tid] = r;
        }
    }
    __syncthreads();

    const int wave = tid >> 5;
    const int lane = tid & 31;

    v8i acc = {};
    const uint32_t* aptr = apack + (size_t)wave * kSteps * 256 + lane * 8;
    for (int ks = 0; ks < kSteps; ++ks) {
        v8i afrag = *reinterpret_cast<const v8i*>(aptr + ks * 256);
        v8i bfrag = *reinterpret_cast<const v8i*>(&ldsB[ks * 256 + lane * 8]);
        acc = __builtin_amdgcn_wmma_i32_16x16x64_iu8(
             false, afrag,  false, bfrag, acc,
             false,  false);
    }

    {
        int n   = lane & 15;
        int p   = ntile * 16 + n;
        int bb  = p / kHW;
        int rem = p - bb * kHW;
        int h   = rem / kW;
        int w   = rem - h * kW;
        int mhi = (lane & 16) ? 8 : 0;
#pragma unroll
        for (int r = 0; r < 8; ++r) {
            int cout = wave * 16 + r + mhi;
            size_t oidx = (((size_t)bb * kCout + cout) * kH + h) * kW + w;
            out[oidx] = (float)acc[r] + bias[cout];
        }
    }
}

extern "C" void kernel_launch(void* const* d_in, const int* in_sizes, int n_in,
                              void* d_out, int out_size, void* d_ws, size_t ws_size,
                              hipStream_t stream) {
    const float* x     = (const float*)d_in[0];
    const float* pcilt = (const float*)d_in[1];
    const float* bias  = (const float*)d_in[2];
    float* out = (float*)d_out;

    uint8_t*  xq    = (uint8_t*)d_ws;
    uint32_t* apack = (uint32_t*)((char*)d_ws + (size_t)kB * kCin * kHW);

    int n4 = (kB * kCin * kHW) / 4;
    quant_x_kernel<<<(n4 + 255) / 256, 256, 0, stream>>>(x, (uint32_t*)xq, n4);

    int wdwords = kMTiles * kSteps * 32 * 8;
    pack_w_kernel<<<(wdwords + 255) / 256, 256, 0, stream>>>(pcilt, apack);

    conv_wmma_kernel<<<kNpix / 16, 256, 0, stream>>>(xq, apack, bias, out);
}
